// SelfAttention_13950053778190
// MI455X (gfx1250) — compile-verified, matrix operand order checked by reading, not run on hardware
//
#include <hip/hip_runtime.h>

typedef __attribute__((ext_vector_type(16))) __bf16 v16bf;
typedef __attribute__((ext_vector_type(8)))  float  v8f;

namespace {
constexpr int kB   = 8;
constexpr int kC   = 256;
constexpr int kCqk = 32;
constexpr int kN   = 4096;
constexpr int kPad = 36;   // padded LDS row (bf16 elems): bank-conflict-free

// workspace layout (bytes)
constexpr size_t OFF_SUMSQ = 0;                                    // 8*64 f32
constexpr size_t OFF_BIAS  = 4096;                                 // 320 f32
constexpr size_t OFF_WCAT  = 8192;                                 // 320x256 bf16
constexpr size_t OFF_XT    = 256 * 1024;                           // B x N x C bf16
constexpr size_t OFF_QF    = OFF_XT + (size_t)kB * kN * kC * 2;    // B x N x 32 f32
constexpr size_t OFF_KF    = OFF_QF + (size_t)kB * kN * kCqk * 4;  // B x N x 32 f32
constexpr size_t OFF_QB    = OFF_KF + (size_t)kB * kN * kCqk * 4;  // B x N x 32 bf16
constexpr size_t OFF_KB    = OFF_QB + (size_t)kB * kN * kCqk * 2;  // B x N x 32 bf16
constexpr size_t OFF_VT    = OFF_KB + (size_t)kB * kN * kCqk * 2;  // B x N x C bf16
constexpr size_t OFF_VCN   = OFF_VT + (size_t)kB * kN * kC * 2;    // B x C x N bf16
}

union frag_u { v16bf v; unsigned long long q[4]; __bf16 h[16]; };

__device__ __forceinline__ int lane_id() { return (int)(threadIdx.x & 31); }

// A-frag 16x32 from row-major (m,k) memory; equally a B-frag 32x16 when the
// B matrix is stored n-major (element (k,n) at base[n*ld + k]).
// Two contiguous b64 loads per half-wave; matches ISA 7.12.2 striping.
__device__ __forceinline__ v16bf load_frag_nmajor(const __bf16* base, int ld) {
  int l = lane_id();
  int r = l & 15;
  int kb = (l & 16) ? 8 : 0;
  const unsigned long long* p = (const unsigned long long*)(base + (size_t)r * ld);
  frag_u u;
  u.q[0] = p[(kb >> 2) + 0];          // k = kb .. kb+3
  u.q[1] = p[(kb >> 2) + 1];          // k = kb+4 .. kb+7
  u.q[2] = p[((kb + 16) >> 2) + 0];   // k = kb+16 .. kb+19
  u.q[3] = p[((kb + 16) >> 2) + 1];   // k = kb+20 .. kb+23
  return u.v;
}

__device__ __forceinline__ v8f wmma_bf16(v16bf a, v16bf b, v8f c) {
  return __builtin_amdgcn_wmma_f32_16x16x32_bf16(false, a, false, b, (short)0, c,
                                                 false, false);
}

__device__ __forceinline__ void async_b128(unsigned lds_addr, const void* g) {
  asm volatile("global_load_async_to_lds_b128 %0, %1, off"
               :: "v"(lds_addr), "v"((unsigned long long)(size_t)g) : "memory");
}
__device__ __forceinline__ void wait_async0() {
  asm volatile("s_wait_asynccnt 0x0" ::: "memory");
}
__device__ __forceinline__ void wait_ds0() {
  asm volatile("s_wait_dscnt 0x0" ::: "memory");
}

__device__ __forceinline__ float hmax16(float v) {
#pragma unroll
  for (int m = 1; m <= 8; m <<= 1) v = fmaxf(v, __shfl_xor(v, m, 32));
  return v;
}
__device__ __forceinline__ float hsum16(float v) {
#pragma unroll
  for (int m = 1; m <= 8; m <<= 1) v += __shfl_xor(v, m, 32);
  return v;
}

// ---------------------------------------------------------------- K0: weights
// Wcat (320 x 256 row-major bf16) = [Wq ; Wk ; Wv] -- already n-major for the
// projection GEMM's B operand, so no transpose and no strided loads anywhere.
__global__ void init_wt(const float* Wq, const float* Wk, const float* Wv,
                        const float* bq, const float* bk, const float* bv,
                        __bf16* Wcat, float* bias, float* sumsq) {
  int i = blockIdx.x * 256 + threadIdx.x;
  if (i < kB * 64) sumsq[i] = 0.0f;
  if (i < 320) bias[i] = (i < 32) ? bq[i] : (i < 64) ? bk[i - 32] : bv[i - 64];
  if (i < 320 * 256) {
    int j = i / 256, c = i % 256;   // j = output column, c = reduction index
    float w = (j < 32)   ? Wq[j * 256 + c]
              : (j < 64) ? Wk[(j - 32) * 256 + c]
                         : Wv[(j - 64) * 256 + c];
    Wcat[(size_t)j * 256 + c] = (__bf16)w;
  }
}

// --------------------------------------------------- K1: x (C,N) -> Xt (N,C)
__global__ void cast_transpose_x(const float* x, __bf16* Xt) {
  __shared__ float tile[32][33];
  int b = blockIdx.z, n0 = blockIdx.x * 32, c0 = blockIdx.y * 32;
  int tx = threadIdx.x, ty = threadIdx.y;
#pragma unroll
  for (int i = 0; i < 4; ++i)
    tile[ty + i * 8][tx] =
        x[((size_t)b * kC + (c0 + ty + i * 8)) * kN + n0 + tx];
  __syncthreads();
#pragma unroll
  for (int i = 0; i < 4; ++i)
    Xt[((size_t)b * kN + (n0 + ty + i * 8)) * kC + c0 + tx] =
        (__bf16)tile[tx][ty + i * 8];
}

// ------------------------------------------- K2: Y(N x 320) = Xt * W^T + bias
__global__ __launch_bounds__(256) void proj_gemm(
    const __bf16* Xt, const __bf16* Wcat, const float* biasc, float* Qf,
    float* Kf, __bf16* Vt, float* sumsq) {
  int b = blockIdx.z;
  int wave = threadIdx.x >> 5, lane = lane_id();
  int m0 = blockIdx.x * 128 + wave * 16;
  int c0 = blockIdx.y * 64;
  const __bf16* X = Xt + ((size_t)b * kN + m0) * kC;
  v8f acc[4] = {};
  for (int kk = 0; kk < kC; kk += 32) {
    v16bf a = load_frag_nmajor(X + kk, kC);
#pragma unroll
    for (int f = 0; f < 4; ++f) {
      v16bf bf = load_frag_nmajor(Wcat + (size_t)(c0 + f * 16) * kC + kk, kC);
      acc[f] = wmma_bf16(a, bf, acc[f]);
    }
  }
  int h8 = (lane & 16) ? 8 : 0;
#pragma unroll
  for (int f = 0; f < 4; ++f) {
    int col = c0 + f * 16 + (lane & 15);
    float bia = biasc[col];
    float ss = 0.0f;
#pragma unroll
    for (int r = 0; r < 8; ++r) {
      float val = acc[f][r] + bia;
      int n = m0 + r + h8;
      if (col < kCqk) {
        Qf[((size_t)b * kN + n) * kCqk + col] = val;
        ss += val * val;
      } else if (col < 2 * kCqk) {
        Kf[((size_t)b * kN + n) * kCqk + (col - kCqk)] = val;
        ss += val * val;
      } else {
        Vt[((size_t)b * kN + n) * kC + (col - 64)] = (__bf16)val;
      }
    }
    if (col < 64) atomicAdd(&sumsq[b * 64 + col], ss);
  }
}

// ------------------------- K3: fold l2-norm (+ temperature into Q), cast bf16
__global__ void normalize_qk(const float* Qf, const float* Kf,
                             const float* sumsq, const float* temp_p,
                             __bf16* Qb, __bf16* Kb) {
  size_t i = (size_t)blockIdx.x * 256 + threadIdx.x;  // over B*N*32
  int c = (int)(i & 31);
  int b = (int)(i >> 17);  // N*32 == 1<<17
  float invt = 1.0f / (temp_p[0] + 1e-6f);
  float sq = invt / fmaxf(sqrtf(sumsq[b * 64 + c]), 1e-12f);
  float sk = 1.0f / fmaxf(sqrtf(sumsq[b * 64 + 32 + c]), 1e-12f);
  Qb[i] = (__bf16)(Qf[i] * sq);
  Kb[i] = (__bf16)(Kf[i] * sk);
}

// ----------------------------------------------- K3b: Vt (N,C) -> Vcn (C,N)
__global__ void transpose_v(const __bf16* Vt, __bf16* Vcn) {
  __shared__ __bf16 tile[32][33];
  int b = blockIdx.z, n0 = blockIdx.x * 32, c0 = blockIdx.y * 32;
  int tx = threadIdx.x, ty = threadIdx.y;
#pragma unroll
  for (int i = 0; i < 4; ++i)
    tile[ty + i * 8][tx] =
        Vt[((size_t)b * kN + (n0 + ty + i * 8)) * kC + c0 + tx];
  __syncthreads();
#pragma unroll
  for (int i = 0; i < 4; ++i)
    Vcn[((size_t)b * kC + (c0 + ty + i * 8)) * kN + n0 + tx] =
        tile[tx][ty + i * 8];
}

// -------------------------------------------------- K4: fused flash attention
// Per wave: 16 query rows.  S = Q K^T (2 WMMA) -> online softmax -> Ot += V P^T
// (16 WMMA; accumulator is O-transposed so every fragment load is contiguous).
// LDS rows padded to 36 elems (72 B): bank = (18*r) mod 64 is distinct for all
// 16 lanes -> conflict-free ds_read_b64 feeds for the P*V WMMAs.
__global__ __launch_bounds__(256) void flash_attn(
    const __bf16* Qb, const __bf16* Kb, const __bf16* Vcn, const float* xin,
    const float* gammap, float* out) {
  __shared__ __align__(16) __bf16 ldsV[kC * kPad];     // V tile (C x 32 keys)
  __shared__ __align__(16) __bf16 Pbuf[8][16 * kPad];  // per-wave P transpose
  __shared__ float Abuf[8][16];
  __shared__ float Lbuf[8][16];

  int b = blockIdx.y;
  int wave = threadIdx.x >> 5, lane = lane_id();
  int tid = threadIdx.x;
  int m0 = blockIdx.x * 128 + wave * 16;
  int h8 = (lane & 16) ? 8 : 0;

  const __bf16* Q = Qb + ((size_t)b * kN + m0) * kCqk;
  const __bf16* K = Kb + (size_t)b * kN * kCqk;
  const __bf16* V = Vcn + (size_t)b * kC * kN;

  v16bf qf = load_frag_nmajor(Q, kCqk);
  v8f o[16] = {};
  float mrow[8], lrow[8];
#pragma unroll
  for (int r = 0; r < 8; ++r) { mrow[r] = -1e30f; lrow[r] = 0.0f; }

  unsigned ldsVbase = (unsigned)(size_t)(&ldsV[0]);
  v8f zf = {};

  for (int nb = 0; nb < kN / 32; ++nb) {
    int n0 = nb * 32;
    // --- async stage V tile (contiguous 64B of data per padded 72B row)
    {
      const __bf16* g = V + (size_t)tid * kN + n0;
      unsigned l = ldsVbase + (unsigned)tid * (kPad * 2u);
#pragma unroll
      for (int j = 0; j < 4; ++j) async_b128(l + j * 16u, g + j * 8);
    }
    // --- S = Q * K^T  (K tile is a contiguous 2KB block; hits WGP$/L2)
    v16bf k0 = load_frag_nmajor(K + (size_t)n0 * kCqk, kCqk);
    v16bf k1 = load_frag_nmajor(K + (size_t)(n0 + 16) * kCqk, kCqk);
    v8f s0 = wmma_bf16(qf, k0, zf);
    v8f s1 = wmma_bf16(qf, k1, zf);
    // --- + identity
#pragma unroll
    for (int r = 0; r < 8; ++r) {
      int t = (m0 + r + h8) - n0;
      if (t >= 0 && t < 16 && (lane & 15) == t) s0[r] += 1.0f;
      if (t >= 16 && t < 32 && (lane & 15) == (t - 16)) s1[r] += 1.0f;
    }
    // --- online softmax
#pragma unroll
    for (int r = 0; r < 8; ++r) {
      float mx = hmax16(fmaxf(s0[r], s1[r]));
      float mnew = fmaxf(mrow[r], mx);
      float al = __expf(mrow[r] - mnew);
      mrow[r] = mnew;
      float e0 = __expf(s0[r] - mnew);
      float e1 = __expf(s1[r] - mnew);
      lrow[r] = lrow[r] * al + hsum16(e0 + e1);
      if ((lane & 15) == 0) Abuf[wave][r + h8] = al;
      Pbuf[wave][(r + h8) * kPad + (lane & 15)] = (__bf16)e0;
      Pbuf[wave][(r + h8) * kPad + 16 + (lane & 15)] = (__bf16)e1;
    }
    wait_ds0();  // own-wave LDS RAW
    float av = Abuf[wave][lane & 15];
    v16bf pB = load_frag_nmajor(&Pbuf[wave][0], kPad);  // B = P^T (reused 16x)
#pragma unroll
    for (int f = 0; f < 16; ++f) o[f] = o[f] * av;      // rescale by alpha
    // --- wait for V tile, then Ot += V * P^T
    wait_async0();
    __syncthreads();
#pragma unroll
    for (int f = 0; f < 16; ++f) {
      v16bf vA = load_frag_nmajor(ldsV + f * 16 * kPad, kPad);
      o[f] = wmma_bf16(vA, pB, o[f]);
    }
    __syncthreads();  // everyone done reading ldsV before next overwrite
  }

  // --- epilogue: 1/l per query column, gamma * O + x
  if ((lane & 15) == 0) {
#pragma unroll
    for (int r = 0; r < 8; ++r) Lbuf[wave][r + h8] = 1.0f / lrow[r];
  }
  wait_ds0();
  float linv = Lbuf[wave][lane & 15];
  float g = gammap[0];
  int m = m0 + (lane & 15);
#pragma unroll
  for (int f = 0; f < 16; ++f) {
#pragma unroll
    for (int r = 0; r < 8; ++r) {
      int c = f * 16 + r + h8;
      size_t idx = ((size_t)b * kC + c) * kN + m;
      out[idx] = g * (o[f][r] * linv) + xin[idx];
    }
  }
}

extern "C" void kernel_launch(void* const* d_in, const int* in_sizes, int n_in,
                              void* d_out, int out_size, void* d_ws,
                              size_t ws_size, hipStream_t stream) {
  (void)in_sizes; (void)n_in; (void)out_size; (void)ws_size;
  const float* x     = (const float*)d_in[0];
  const float* Wq    = (const float*)d_in[1];
  const float* bq    = (const float*)d_in[2];
  const float* Wk    = (const float*)d_in[3];
  const float* bk    = (const float*)d_in[4];
  const float* Wv    = (const float*)d_in[5];
  const float* bv    = (const float*)d_in[6];
  const float* gamma = (const float*)d_in[7];
  const float* temp  = (const float*)d_in[8];
  float* out = (float*)d_out;

  char* ws = (char*)d_ws;
  float* sumsq = (float*)(ws + OFF_SUMSQ);
  float* biasc = (float*)(ws + OFF_BIAS);
  __bf16* Wcat = (__bf16*)(ws + OFF_WCAT);
  __bf16* Xt   = (__bf16*)(ws + OFF_XT);
  float* Qf    = (float*)(ws + OFF_QF);
  float* Kf    = (float*)(ws + OFF_KF);
  __bf16* Qb   = (__bf16*)(ws + OFF_QB);
  __bf16* Kb   = (__bf16*)(ws + OFF_KB);
  __bf16* Vt   = (__bf16*)(ws + OFF_VT);
  __bf16* Vcn  = (__bf16*)(ws + OFF_VCN);

  init_wt<<<320, 256, 0, stream>>>(Wq, Wk, Wv, bq, bk, bv, Wcat, biasc, sumsq);
  cast_transpose_x<<<dim3(kN / 32, kC / 32, kB), dim3(32, 8), 0, stream>>>(x, Xt);
  proj_gemm<<<dim3(kN / 128, 320 / 64, kB), 256, 0, stream>>>(Xt, Wcat, biasc,
                                                              Qf, Kf, Vt, sumsq);
  normalize_qk<<<(kB * kN * kCqk) / 256, 256, 0, stream>>>(Qf, Kf, sumsq, temp,
                                                           Qb, Kb);
  transpose_v<<<dim3(kN / 32, kC / 32, kB), dim3(32, 8), 0, stream>>>(Vt, Vcn);
  flash_attn<<<dim3(kN / 128, kB), 256, 0, stream>>>(Qb, Kb, Vcn, x, gamma, out);
}
